// SelectiveDense_61942018342944
// MI455X (gfx1250) — matrix operand order repaired and checked by reading, not compiled, not run on hardware
//
#include <hip/hip_runtime.h>
#include <hip/hip_bf16.h>

// ---------------------------------------------------------------------------
// SelectiveDense for MI455X (gfx1250, wave32, WMMA).
//   D=32, IN=1024, P=528 -> padded to 36 tiles of 16 (no EXEC guards in loop)
//   Kernel 1: W = g * V / ||V||_row -> bf16, pre-swizzled to WMMA B-fragment
//   Kernel 2: fused  X@W^T (v_wmma_f32_16x16x32_bf16, A staged in LDS)
//             + triu scatter + Gram QQ^T + 32-step power iter + apply
// ---------------------------------------------------------------------------

#define D_   32
#define IN_  1024
#define P_   528          // true P (33 * 16)
#define NTP_ 36           // padded tile count: 9 tiles per wave, uniform
#define SP_  576          // padded S pitch = NTP_ * 16
#define KH_  512          // K half staged in LDS per pass
#define EPS_ 1e-5f

typedef __attribute__((ext_vector_type(16))) __bf16 v16bf_t;
typedef __attribute__((ext_vector_type(8)))  float  v8f_t;

struct alignas(8) BF4 { __bf16 v[4]; };

// A-fragment index (16-bit A, 16x32 per k-step, wave32), for element A[m][k]:
//   lane L = 16h + m ; VGPR pair j ; element 2j+e
//   k%32:  0-7 -> h0 j0-3 | 8-15 -> h1 j0-3 | 16-23 -> h0 j4-7 | 24-31 -> h1 j4-7
__device__ __forceinline__ int afrag_index(int m, int klocal) {
    int s  = klocal >> 5;
    int r2 = klocal & 31;
    int h  = (r2 >> 3) & 1;
    int j  = ((r2 >> 4) << 2) + ((r2 & 7) >> 1);
    int L  = h * 16 + m;
    return (s * 32 + L) * 16 + 2 * j + (klocal & 1);
}

// ---------------------------------------------------------------------------
// Kernel 1: weight-norm + bf16 swizzle into WMMA B-fragment order
//   B-fragment (16-bit, 32x16, wave32): lane L: col n=L%16, K-half h=L/16,
//   VGPR j: K = 32s + 16(j>>2) + 8h + 2(j&3) + {0,1}.  32B/lane contiguous.
// ---------------------------------------------------------------------------
__global__ __launch_bounds__(256)
void prep_w_kernel(const float* __restrict__ V, const float* __restrict__ g,
                   __bf16* __restrict__ WF) {
    const int p   = blockIdx.x;        // 0..527
    const int tid = threadIdx.x;       // 0..255
    __shared__ float red[256];

    const float* vr = V + (size_t)p * IN_;
    float ss = 0.f;
#pragma unroll
    for (int r = 0; r < 4; ++r) {
        float x = vr[tid + 256 * r];
        ss += x * x;
    }
    red[tid] = ss;
    __syncthreads();
    for (int off = 128; off > 0; off >>= 1) {
        if (tid < off) red[tid] += red[tid + off];
        __syncthreads();
    }
    const float scale = g[p] * rsqrtf(red[0]);   // g / ||V||_row

    const int t = p >> 4;
    const int c = p & 15;
#pragma unroll
    for (int r = 0; r < 4; ++r) {
        int k  = tid + 256 * r;
        float val = vr[k] * scale;
        int s5 = k >> 5;
        int r2 = k & 31;
        int h  = (r2 >> 3) & 1;
        int j  = ((r2 >> 4) << 2) + ((r2 & 7) >> 1);
        int e  = k & 1;
        int L  = h * 16 + c;
        size_t addr = ((size_t)(t * 32 + s5) * 32 + L) * 16 + j * 2 + e;
        WF[addr] = (__bf16)val;
    }
}

// ---------------------------------------------------------------------------
// Kernel 2: fused GEMM + per-token spectral normalization.
//   Grid: N/16 workgroups, 128 threads (4 waves).
// ---------------------------------------------------------------------------
__global__ __launch_bounds__(128)
void fused_kernel(const float* __restrict__ A,      // [N, 1024] param_arg
                  const float* __restrict__ X,      // [N, 32]   input_arg
                  const __bf16* __restrict__ WF,    // swizzled bf16 weights
                  float* __restrict__ out) {        // [N, 32]
    __shared__ float S[16 * SP_];                      // 36.9 KB (persistent)
    __shared__ __align__(32) __bf16 Abf[16 * KH_];     // 16 KB, A frag staging
    __shared__ float vS[16 * D_];                      // 2 KB
    __shared__ float xS[16 * D_];                      // 2 KB

    const int tid   = threadIdx.x;
    const int wave  = __builtin_amdgcn_readfirstlane(tid >> 5);  // scalar
    const int L     = tid & 31;
    const int lan16 = L & 15;
    const int h     = L >> 4;
    const size_t tokBase = (size_t)blockIdx.x * 16;

    // ---------------- Phase 1: S = A_tile @ W^T via WMMA bf16 ----------------
    v8f_t acc[9];
    {
        v8f_t zero = {};
#pragma unroll
        for (int ti = 0; ti < 9; ++ti) acc[ti] = zero;
    }

    for (int half = 0; half < 2; ++half) {
        // --- Phase 0: stage A K-half into LDS, bf16, fragment-swizzled ---
        // thread: row m = tid/8, 64 contiguous floats starting at (tid%8)*64
        {
            const int m   = tid >> 3;
            const int kb0 = (tid & 7) * 64;
            const float* Arow = A + (tokBase + (size_t)m) * IN_ + half * KH_ + kb0;
#pragma unroll
            for (int c = 0; c < 16; ++c) {
                float4 u = *(const float4*)(Arow + c * 4);
                int idx = afrag_index(m, kb0 + c * 4);   // 4 consecutive bf16, 8B aligned
                BF4 tmp;
                tmp.v[0] = (__bf16)u.x; tmp.v[1] = (__bf16)u.y;
                tmp.v[2] = (__bf16)u.z; tmp.v[3] = (__bf16)u.w;
                *(BF4*)(Abf + idx) = tmp;
            }
        }
        __syncthreads();

        // --- WMMA: each wave owns 9 tiles t = wave + 4*ti (uniform, no guard)
#pragma unroll 2
        for (int s = 0; s < 16; ++s) {
            const v16bf_t af = *(const v16bf_t*)(Abf + (s * 32 + L) * 16);
            if (s < 15)  // prefetch next k-step of this wave's first B tile
                __builtin_prefetch(
                    (const void*)(WF + ((size_t)(wave * 32 + half * 16 + s + 1) * 32 + L) * 16),
                    0, 1);
#pragma unroll
            for (int ti = 0; ti < 9; ++ti) {
                const int t = wave + ti * 4;             // 0..35, all valid (padded)
                const v16bf_t bf = *(const v16bf_t*)
                    (WF + ((size_t)(t * 32 + half * 16 + s) * 32 + L) * 16);
                acc[ti] = __builtin_amdgcn_wmma_f32_16x16x32_bf16(
                    false, af, false, bf, (short)0, acc[ti], false, false);
            }
        }
        __syncthreads();   // Abf reused next half
    }

    // C/D layout: VGPR r -> row M = r + 8h, col N = L%16
#pragma unroll
    for (int ti = 0; ti < 9; ++ti) {
        const int t = wave + ti * 4;
        const int colBase = t * 16 + lan16;              // cols >= 528 are dead pad
#pragma unroll
        for (int r = 0; r < 8; ++r)
            S[(r + 8 * h) * SP_ + colBase] = acc[ti][r];
    }
    __syncthreads();

    // ---------------- Phase 2: per-token Gram + power iteration --------------
    // 8 lanes per token (4 tokens per wave); each lane owns 4 rows.
    const int tt  = tid >> 3;       // token 0..15
    const int sub = tid & 7;
    const int i0  = sub * 4;
    const float* Srow = &S[tt * SP_];

    // triu flat index: (i,k) -> ci + k, ci = 32i - i(i-1)/2 - i  (all < 528)
    float G[4][32];                 // Gram rows in registers
#pragma unroll
    for (int r = 0; r < 4; ++r) {
        const int i  = i0 + r;
        const int ci = i * 32 - (i * (i - 1)) / 2 - i;
#pragma unroll
        for (int j = 0; j < 32; ++j) {
            const int cj = j * 32 - (j * (j - 1)) / 2 - j;
            float a = 0.f;
#pragma unroll
            for (int k = j; k < 32; ++k) {        // q(j,k)=0 for k<j
                float qj = Srow[cj + k];
                float qi = (k >= i) ? Srow[ci + k] : 0.f;
                a += qi * qj;
            }
            G[r][j] = a;
        }
    }

    // init v = 1/sqrt(32)
#pragma unroll
    for (int r = 0; r < 4; ++r)
        vS[tt * D_ + i0 + r] = 0.17677669529663687f;
    __syncthreads();

    float eig = 0.f;
    for (int it = 0; it < 32; ++it) {
        float u[4];
#pragma unroll
        for (int r = 0; r < 4; ++r) {
            float a = 0.f;
#pragma unroll
            for (int j = 0; j < 32; ++j) a += G[r][j] * vS[tt * D_ + j];
            u[r] = a;
        }
        float nrm = u[0]*u[0] + u[1]*u[1] + u[2]*u[2] + u[3]*u[3];
        nrm += __shfl_xor(nrm, 1, 32);     // reduce across the 8-lane group
        nrm += __shfl_xor(nrm, 2, 32);
        nrm += __shfl_xor(nrm, 4, 32);
        eig = sqrtf(nrm);
        const float inv = 1.f / (eig + EPS_);
        __syncthreads();
#pragma unroll
        for (int r = 0; r < 4; ++r)
            vS[tt * D_ + i0 + r] = u[r] * inv;
        __syncthreads();
    }

    // stage input vectors and apply G/(eig+eps)
#pragma unroll
    for (int r = 0; r < 4; ++r) {
        int q = tid + 128 * r;
        xS[q] = X[tokBase * D_ + q];
    }
    __syncthreads();

    const float inv = 1.f / (eig + EPS_);
#pragma unroll
    for (int r = 0; r < 4; ++r) {
        float a = 0.f;
#pragma unroll
        for (int j = 0; j < 32; ++j) a += G[r][j] * xS[tt * D_ + j];
        out[(tokBase + tt) * D_ + i0 + r] = a * inv;
    }
}

// ---------------------------------------------------------------------------
extern "C" void kernel_launch(void* const* d_in, const int* in_sizes, int n_in,
                              void* d_out, int out_size, void* d_ws, size_t ws_size,
                              hipStream_t stream) {
    const float* param = (const float*)d_in[0];   // [B,T,1024]
    const float* input = (const float*)d_in[1];   // [B,T,32]
    const float* vwt   = (const float*)d_in[2];   // [528,1024]
    const float* g     = (const float*)d_in[3];   // [528]
    float* out = (float*)d_out;
    __bf16* WF = (__bf16*)d_ws;                   // 36*32*512 bf16 ~= 1.13 MB

    const int N = in_sizes[1] / D_;               // 32768 tokens

    prep_w_kernel<<<P_, 256, 0, stream>>>(vwt, g, WF);
    fused_kernel<<<N / 16, 128, 0, stream>>>(param, input, WF, out);
}
